// TransformerModel_24318104830541
// MI455X (gfx1250) — matrix operand order repaired and checked by reading, not compiled, not run on hardware
//
#include <hip/hip_runtime.h>

// ---------- problem constants (match reference) ----------
#define NL   6
#define HH   16
#define DD   1024
#define DKK  64
#define DFF_ 4096
#define SS   2048
#define BB   4
#define MTOK (BB * SS)   // 8192 rows of tokens

typedef __attribute__((ext_vector_type(16))) __bf16 v16bf;
typedef __attribute__((ext_vector_type(8)))  float  v8f;
typedef __attribute__((ext_vector_type(4)))  unsigned int uv4;   // 16B raw chunk (trivially copyable)
typedef __attribute__((ext_vector_type(4)))  float  fv4;

union BFrag {            // 16 bf16 = 32B = one WMMA A/B operand (8 VGPRs)
    uv4   u[2];
    v16bf v;
};

__device__ __forceinline__ v8f wmma_bf16(const v16bf& a, const v16bf& b, v8f c) {
    return __builtin_amdgcn_wmma_f32_16x16x32_bf16(false, a, false, b, (short)0, c, false, false);
}

// CDNA5 async copy: 16 bytes per lane, global -> LDS, tracked by ASYNCcnt.
// lds generic pointer's low 32 bits are the wave-relative LDS byte address.
__device__ __forceinline__ void async_b128(const void* lds, const void* g) {
    asm volatile("global_load_async_to_lds_b128 %0, %1, off"
                 :: "v"((unsigned)(unsigned long long)lds), "v"(g)
                 : "memory");
}
__device__ __forceinline__ void wait_async_le3() {
    asm volatile("s_wait_asynccnt 0x3" ::: "memory");
}
__device__ __forceinline__ void wait_async_0() {
    asm volatile("s_wait_asynccnt 0x0" ::: "memory");
}

// =====================================================================
// Embedding + positional encoding:  x = emb[tgt]*sqrt(D) + pe
// =====================================================================
__global__ __launch_bounds__(256) void embed_kernel(const long long* __restrict__ tgt,
                                                    const float* __restrict__ emb,
                                                    const float* __restrict__ pe,
                                                    float* __restrict__ x) {
    size_t row  = blockIdx.x;                 // b*S + s
    int    srow = (int)(row & (SS - 1));
    long long id = tgt[row];
    int c = threadIdx.x * 4;
    fv4 e = *(const fv4*)(emb + (size_t)id * DD + c);
    fv4 p = *(const fv4*)(pe  + (size_t)srow * DD + c);
    fv4 o;
    o.x = e.x * 32.0f + p.x;  o.y = e.y * 32.0f + p.y;
    o.z = e.z * 32.0f + p.z;  o.w = e.w * 32.0f + p.w;
    *(fv4*)(x + row * DD + c) = o;
}

// =====================================================================
// LayerNorm (unbiased var, torch-style a*(x-m)/(std+eps)+b)
// =====================================================================
__device__ __forceinline__ float block_sum(float v, float* red) {
    #pragma unroll
    for (int o = 16; o > 0; o >>= 1) v += __shfl_xor(v, o, 32);
    if ((threadIdx.x & 31) == 0) red[threadIdx.x >> 5] = v;
    __syncthreads();
    float tot = 0.f;
    #pragma unroll
    for (int i = 0; i < 8; ++i) tot += red[i];
    __syncthreads();
    return tot;
}

template<bool OUTF>
__global__ __launch_bounds__(256) void ln_kernel(const float* __restrict__ x,
                                                 const float* __restrict__ ga,
                                                 const float* __restrict__ be,
                                                 __bf16* __restrict__ ob,
                                                 float* __restrict__ of) {
    __shared__ float red[8];
    size_t row = blockIdx.x;
    const float* xr = x + row * (size_t)DD;
    int c = threadIdx.x * 4;
    fv4 xv = *(const fv4*)(xr + c);
    float s = xv.x + xv.y + xv.z + xv.w;
    s = block_sum(s, red);
    float mean = s * (1.0f / DD);
    float d0 = xv.x - mean, d1 = xv.y - mean, d2 = xv.z - mean, d3 = xv.w - mean;
    float sq = d0*d0 + d1*d1 + d2*d2 + d3*d3;
    sq = block_sum(sq, red);
    float inv = 1.0f / (sqrtf(sq * (1.0f / (DD - 1))) + 1e-6f);
    fv4 g = *(const fv4*)(ga + c);
    fv4 b = *(const fv4*)(be + c);
    float r0 = g.x * d0 * inv + b.x;
    float r1 = g.y * d1 * inv + b.y;
    float r2 = g.z * d2 * inv + b.z;
    float r3 = g.w * d3 * inv + b.w;
    if (OUTF) {
        fv4 o; o.x = r0; o.y = r1; o.z = r2; o.w = r3;
        *(fv4*)(of + row * DD + c) = o;
    } else {
        __bf16* p = ob + row * DD + c;
        p[0] = (__bf16)r0; p[1] = (__bf16)r1; p[2] = (__bf16)r2; p[3] = (__bf16)r3;
    }
}

// =====================================================================
// Weight packing: fp32 -> bf16, TRANSPOSED to [N][K] so GEMM B-staging is
// a straight contiguous copy (async-to-LDS friendly).
// =====================================================================
__global__ __launch_bounds__(256) void pack_headw_t(const float* __restrict__ src,  // (H, D, DK)
                                                    __bf16* __restrict__ dst) {     // [H*DK][D]
    int i = blockIdx.x * 256 + threadIdx.x;     // dst index over 1024*1024
    int n = i >> 10, d = i & 1023;              // dst[n*D + d], n = h*64+kk
    int h = n >> 6,  kk = n & 63;
    dst[i] = (__bf16)src[((size_t)h * DD + d) * DKK + kk];
}

__global__ __launch_bounds__(256) void pack_plain_t(const float* __restrict__ src,  // (K, N)
                                                    __bf16* __restrict__ dst,       // [N][K]
                                                    int kshift, int N) {
    int i = blockIdx.x * 256 + threadIdx.x;     // dst index over N*K
    int n = i >> kshift;
    int k = i & ((1 << kshift) - 1);
    dst[i] = (__bf16)src[(size_t)k * N + n];
}

// =====================================================================
// WMMA GEMM:  C[M,N] = A[M,K](bf16) * B^T (Bw is [N][K] bf16) + bias
// WG = 256 thr = 8 waves, tile 128x64, BK=32, double-buffered LDS filled
// with global_load_async_to_lds_b128. Wave w owns rows w*16..w*16+15 and
// all 64 cols: 4 accumulators, 1 A-frag + 4 B-frags = 2.5 ds_load_b128
// per WMMA, one barrier pair per 4 WMMAs per wave.
// =====================================================================
#define BKP 40   // padded k-stride: 80B rows, 16B aligned, conflict-free (r*20 mod 64 is a permutation)

template<bool RELU, bool RES, bool OUTB>
__global__ __launch_bounds__(256) void gemm_wmma(const __bf16* __restrict__ A,
                                                 const __bf16* __restrict__ Bw,
                                                 const float* __restrict__ bias,
                                                 const float* __restrict__ res,
                                                 float* __restrict__ outF,
                                                 __bf16* __restrict__ outB,
                                                 int M, int N, int K) {
    __shared__ __attribute__((aligned(16))) __bf16 As[2][128][BKP];
    __shared__ __attribute__((aligned(16))) __bf16 Bs[2][64][BKP];

    const int mBase = blockIdx.y * 128, nBase = blockIdx.x * 64;
    const int w  = threadIdx.x >> 5;
    const int lane = threadIdx.x & 31;
    const int r  = lane & 15, hi = lane >> 4;

    v8f acc[4];
    #pragma unroll
    for (int s2 = 0; s2 < 4; ++s2)
        #pragma unroll
        for (int j = 0; j < 8; ++j) acc[s2][j] = 0.f;

    // staging map: 16B per thread per chunk; A = 128x32 (2 chunks), B = 64x32 (1 chunk)
    const int srow = threadIdx.x >> 2;          // 0..63
    const int skc  = (threadIdx.x & 3) << 3;

    auto fill = [&](int buf, int kb) {
        async_b128(&As[buf][srow][skc],      A  + (size_t)(mBase + srow) * K + kb + skc);
        async_b128(&As[buf][srow + 64][skc], A  + (size_t)(mBase + srow + 64) * K + kb + skc);
        async_b128(&Bs[buf][srow][skc],      Bw + (size_t)(nBase + srow) * K + kb + skc);
    };

    fill(0, 0);                                  // prologue
    int buf = 0;
    for (int kb = 0; kb < K; kb += 32, buf ^= 1) {
        const bool hasNext = (kb + 32) < K;
        if (hasNext) { fill(buf ^ 1, kb + 32); wait_async_le3(); }
        else         { wait_async_0(); }
        __syncthreads();

        BFrag a;
        a.u[0] = *(const uv4*)&As[buf][w * 16 + r][hi * 8];
        a.u[1] = *(const uv4*)&As[buf][w * 16 + r][16 + hi * 8];
        #pragma unroll
        for (int s2 = 0; s2 < 4; ++s2) {
            BFrag b;
            const __bf16* br = &Bs[buf][s2 * 16 + r][hi * 8];
            b.u[0] = *(const uv4*)br;
            b.u[1] = *(const uv4*)(br + 16);
            acc[s2] = wmma_bf16(a.v, b.v, acc[s2]);
        }
        __syncthreads();                          // readers done before next overwrite
    }

    #pragma unroll
    for (int s2 = 0; s2 < 4; ++s2) {
        const int col = nBase + s2 * 16 + r;
        const float bsv = bias[col];
        #pragma unroll
        for (int j = 0; j < 8; ++j) {
            const int rowg = mBase + w * 16 + hi * 8 + j;
            float v = acc[s2][j] + bsv;
            if (RELU) v = fmaxf(v, 0.f);
            const size_t off = (size_t)rowg * N + col;
            if (RES) v += res[off];
            if (OUTB) outB[off] = (__bf16)v; else outF[off] = v;
        }
    }
}

// =====================================================================
// Flash-style causal attention. WG = 128 thr = 4 waves; each wave owns
// 16 q-rows, workgroup = one (b, h, 64-row q block). q/k/v/o layout:
// [b, s, h, dk] i.e. element ((b*S+s)*D + h*64 + dk), bf16.
// K tile staged via async-to-LDS; V needs a register transpose.
// =====================================================================
__global__ __launch_bounds__(128) void attn_kernel(const __bf16* __restrict__ q,
                                                   const __bf16* __restrict__ k,
                                                   const __bf16* __restrict__ v,
                                                   __bf16* __restrict__ o) {
    __shared__ __attribute__((aligned(16))) __bf16 Kt[64][72];  // [t][dk]  (B-layout for Q*K^T)
    __shared__ __attribute__((aligned(16))) __bf16 Vt[64][72];  // [dk][t]  (B-layout for P*V)
    __shared__ __attribute__((aligned(16))) __bf16 Ps[64][72];  // [q_local][t]

    const int qb = blockIdx.x, h = blockIdx.y, b = blockIdx.z;
    const int tid = threadIdx.x;
    const int w = tid >> 5, lane = tid & 31;
    const int r = lane & 15, hi = lane >> 4;

    // preload this wave's Q fragments (dk split into two 32-wide k-steps)
    const int qrow = qb * 64 + w * 16 + r;
    BFrag qf[2];
    #pragma unroll
    for (int c = 0; c < 2; ++c) {
        const __bf16* qp = q + ((size_t)(b * SS + qrow)) * DD + h * 64 + c * 32;
        qf[c].u[0] = *(const uv4*)(qp + hi * 8);
        qf[c].u[1] = *(const uv4*)(qp + 16 + hi * 8);
    }

    float m_[8], l_[8];
    v8f oa[4];
    #pragma unroll
    for (int j = 0; j < 8; ++j) { m_[j] = -3.0e38f; l_[j] = 0.f; }
    #pragma unroll
    for (int nd = 0; nd < 4; ++nd)
        #pragma unroll
        for (int j = 0; j < 8; ++j) oa[nd][j] = 0.f;

    for (int tb = 0; tb <= qb; ++tb) {           // causal: only tiles with t_base <= q_base
        // cooperative stage: K async copy, V transpose through registers
        #pragma unroll
        for (int it = 0; it < 4; ++it) {
            const int idx  = tid + it * 128;      // 0..511 chunks of 8 bf16
            const int trow = idx >> 3;
            const int dc   = (idx & 7) << 3;
            const size_t g = ((size_t)(b * SS + tb * 64 + trow)) * DD + h * 64 + dc;
            async_b128(&Kt[trow][dc], k + g);
            union { uv4 u; __bf16 e[8]; } uvv;
            uvv.u = *(const uv4*)(v + g);
            #pragma unroll
            for (int i = 0; i < 8; ++i) Vt[dc + i][trow] = uvv.e[i];
        }
        wait_async_0();
        __syncthreads();

        // scores: S = Q * K^T  (4 t-subtiles of 16)
        v8f sa[4];
        #pragma unroll
        for (int nt = 0; nt < 4; ++nt) {
            v8f accs;
            #pragma unroll
            for (int j = 0; j < 8; ++j) accs[j] = 0.f;
            #pragma unroll
            for (int c = 0; c < 2; ++c) {
                BFrag bf;
                const __bf16* kr = &Kt[nt * 16 + r][c * 32 + hi * 8];
                bf.u[0] = *(const uv4*)kr;
                bf.u[1] = *(const uv4*)(kr + 16);
                accs = wmma_bf16(qf[c].v, bf.v, accs);
            }
            sa[nt] = accs;
        }

        // online softmax per row (rows = hi*8 + j, cols striped over 16 lanes)
        const float sinv = 0.125f;               // 1/sqrt(64)
        #pragma unroll
        for (int j = 0; j < 8; ++j) {
            const int qi = qb * 64 + w * 16 + hi * 8 + j;
            float mt = -3.0e38f;
            #pragma unroll
            for (int nt = 0; nt < 4; ++nt) {
                float val = sa[nt][j] * sinv;
                const int tg = tb * 64 + nt * 16 + r;
                if (tg > qi) val = -1.0e9f;
                sa[nt][j] = val;
                mt = fmaxf(mt, val);
            }
            #pragma unroll
            for (int off = 8; off > 0; off >>= 1) mt = fmaxf(mt, __shfl_xor(mt, off, 32));
            const float mn = fmaxf(m_[j], mt);
            const float alpha = __expf(m_[j] - mn);
            float rs = 0.f;
            #pragma unroll
            for (int nt = 0; nt < 4; ++nt) {
                const float p = __expf(sa[nt][j] - mn);
                sa[nt][j] = p;
                rs += p;
            }
            #pragma unroll
            for (int off = 8; off > 0; off >>= 1) rs += __shfl_xor(rs, off, 32);
            l_[j] = l_[j] * alpha + rs;
            m_[j] = mn;
            #pragma unroll
            for (int nd = 0; nd < 4; ++nd) oa[nd][j] = oa[nd][j] * alpha;
        }

        // spill P (wave-private LDS rows) and reload as A fragments
        #pragma unroll
        for (int nt = 0; nt < 4; ++nt)
            #pragma unroll
            for (int j = 0; j < 8; ++j)
                Ps[w * 16 + hi * 8 + j][nt * 16 + r] = (__bf16)sa[nt][j];
        asm volatile("s_wait_dscnt 0x0" ::: "memory");

        #pragma unroll
        for (int c = 0; c < 2; ++c) {
            BFrag pf;
            const __bf16* pr = &Ps[w * 16 + r][c * 32 + hi * 8];
            pf.u[0] = *(const uv4*)pr;
            pf.u[1] = *(const uv4*)(pr + 16);
            #pragma unroll
            for (int nd = 0; nd < 4; ++nd) {
                BFrag vf;
                const __bf16* vr = &Vt[nd * 16 + r][c * 32 + hi * 8];
                vf.u[0] = *(const uv4*)vr;
                vf.u[1] = *(const uv4*)(vr + 16);
                oa[nd] = wmma_bf16(pf.v, vf.v, oa[nd]);
            }
        }
        __syncthreads();
    }

    // epilogue: o = acc / l, bf16 in [b,s,h,dk]
    #pragma unroll
    for (int nd = 0; nd < 4; ++nd)
        #pragma unroll
        for (int j = 0; j < 8; ++j) {
            const int rowg = qb * 64 + w * 16 + hi * 8 + j;
            const float ov = oa[nd][j] / l_[j];
            o[((size_t)(b * SS + rowg)) * DD + h * 64 + nd * 16 + r] = (__bf16)ov;
        }
}

// =====================================================================
// Host orchestration
// =====================================================================
extern "C" void kernel_launch(void* const* d_in, const int* in_sizes, int n_in,
                              void* d_out, int out_size, void* d_ws, size_t ws_size,
                              hipStream_t stream) {
    (void)in_sizes; (void)n_in; (void)out_size; (void)ws_size;
    const long long* tgt = (const long long*)d_in[0];
    const float* emb = (const float*)d_in[2];
    const float* pe  = (const float*)d_in[3];
    const float* Wq  = (const float*)d_in[4];  const float* bq = (const float*)d_in[5];
    const float* Wk  = (const float*)d_in[6];  const float* bk = (const float*)d_in[7];
    const float* Wv  = (const float*)d_in[8];  const float* bv = (const float*)d_in[9];
    const float* Wo  = (const float*)d_in[10]; const float* bo = (const float*)d_in[11];
    const float* W1  = (const float*)d_in[12]; const float* b1 = (const float*)d_in[13];
    const float* W2  = (const float*)d_in[14]; const float* b2 = (const float*)d_in[15];
    const float* l1a = (const float*)d_in[16]; const float* l1b = (const float*)d_in[17];
    const float* l2a = (const float*)d_in[18]; const float* l2b = (const float*)d_in[19];
    const float* lfa = (const float*)d_in[20]; const float* lfb = (const float*)d_in[21];

    char* p = (char*)d_ws;
    auto take = [&](size_t bytes) { void* r = (void*)p; p += (bytes + 255) & ~(size_t)255; return r; };
    float*  x   = (float*) take((size_t)MTOK * DD * 4);
    __bf16* hln = (__bf16*)take((size_t)MTOK * DD * 2);
    __bf16* qb_ = (__bf16*)take((size_t)MTOK * DD * 2);
    __bf16* kb_ = (__bf16*)take((size_t)MTOK * DD * 2);
    __bf16* vb_ = (__bf16*)take((size_t)MTOK * DD * 2);
    __bf16* ao  = (__bf16*)take((size_t)MTOK * DD * 2);
    __bf16* ff  = (__bf16*)take((size_t)MTOK * DFF_ * 2);
    __bf16* wpk = (__bf16*)take((size_t)DD * DFF_ * 2);

    const dim3 gD (DD / 64,   MTOK / 128);
    const dim3 gF (DFF_ / 64, MTOK / 128);
    const dim3 gAt(SS / 64, HH, BB);

    embed_kernel<<<MTOK, 256, 0, stream>>>(tgt, emb, pe, x);

    for (int l = 0; l < NL; ++l) {
        // --- attention block ---
        ln_kernel<false><<<MTOK, 256, 0, stream>>>(x, l1a + l * DD, l1b + l * DD, hln, nullptr);

        pack_headw_t<<<(DD * DD) / 256, 256, 0, stream>>>(Wq + (size_t)l * HH * DD * DKK, wpk);
        gemm_wmma<false, false, true><<<gD, 256, 0, stream>>>(hln, wpk, bq + l * HH * DKK,
                                                              nullptr, nullptr, qb_, MTOK, DD, DD);
        pack_headw_t<<<(DD * DD) / 256, 256, 0, stream>>>(Wk + (size_t)l * HH * DD * DKK, wpk);
        gemm_wmma<false, false, true><<<gD, 256, 0, stream>>>(hln, wpk, bk + l * HH * DKK,
                                                              nullptr, nullptr, kb_, MTOK, DD, DD);
        pack_headw_t<<<(DD * DD) / 256, 256, 0, stream>>>(Wv + (size_t)l * HH * DD * DKK, wpk);
        gemm_wmma<false, false, true><<<gD, 256, 0, stream>>>(hln, wpk, bv + l * HH * DKK,
                                                              nullptr, nullptr, vb_, MTOK, DD, DD);

        attn_kernel<<<gAt, 128, 0, stream>>>(qb_, kb_, vb_, ao);

        pack_plain_t<<<(DD * DD) / 256, 256, 0, stream>>>(Wo + (size_t)l * DD * DD, wpk, 10, DD);
        gemm_wmma<false, true, false><<<gD, 256, 0, stream>>>(ao, wpk, bo + l * DD,
                                                              x, x, nullptr, MTOK, DD, DD);

        // --- FFN block ---
        ln_kernel<false><<<MTOK, 256, 0, stream>>>(x, l2a + l * DD, l2b + l * DD, hln, nullptr);

        pack_plain_t<<<(DD * DFF_) / 256, 256, 0, stream>>>(W1 + (size_t)l * DD * DFF_, wpk, 10, DFF_);
        gemm_wmma<true, false, true><<<gF, 256, 0, stream>>>(hln, wpk, b1 + l * DFF_,
                                                             nullptr, nullptr, ff, MTOK, DFF_, DD);

        pack_plain_t<<<(DFF_ * DD) / 256, 256, 0, stream>>>(W2 + (size_t)l * DFF_ * DD, wpk, 12, DD);
        gemm_wmma<false, true, false><<<gD, 256, 0, stream>>>(ff, wpk, b2 + l * DD,
                                                              x, x, nullptr, MTOK, DD, DFF_);
    }

    ln_kernel<true><<<MTOK, 256, 0, stream>>>(x, lfa, lfb, nullptr, (float*)d_out);
}
